// Attention_7473243095472
// MI455X (gfx1250) — compile-verified, matrix operand order checked by reading, not run on hardware
//
#include <hip/hip_runtime.h>
#include <hip/hip_bf16.h>
#include <stdint.h>

// ---------------------------------------------------------------------------
// CDNA5 (gfx1250, wave32) attention block:
//   LN -> QKV gemm (WMMA f16) -> flash-attn (WMMA f16, online softmax)
//   -> out-proj gemm (WMMA f16, f32 out + bias)
// Data movement: double-buffered LDS tiles filled with
// GLOBAL_LOAD_ASYNC_TO_LDS_B128 (ASYNCcnt), softmax reductions via DPP8 +
// permlane16 (no LDS bpermute traffic).
// ---------------------------------------------------------------------------

typedef _Float16 v16h __attribute__((ext_vector_type(16)));
typedef float    v8f  __attribute__((ext_vector_type(8)));
typedef uint32_t v4u  __attribute__((ext_vector_type(4)));
typedef int      v4i_vs __attribute__((vector_size(16)));   // builtin param type

#define DIMC   768
#define NSEQ   1024
#define BATCH  8
#define NHEAD  12
#define DHC    64
#define QKVN   2304
#define SCALE_QK 0.125f   // 64^-0.5

union HFrag { v16h h; v4u q[2]; };      // 16 halves = 2 x 16B chunks
union FAcc  { v8f  v; float f[8]; };    // 16x16 f32 C/D tile per lane

// ---------------------------------------------------------------------------
// Async global->LDS 16B copy (ASYNCcnt) with sync fallback.
// AS pointers built via integer casts: generic-LDS low 32 bits == LDS offset,
// generic-global == global address.  Builtin signature (from diagnostics):
//   (AS1 v4i* gsrc, AS3 v4i* ldst, imm offset, imm cpol)
// ---------------------------------------------------------------------------
__device__ __forceinline__ void async_copy16(void* lds, const void* g) {
#if __has_builtin(__builtin_amdgcn_global_load_async_to_lds_b128)
  __builtin_amdgcn_global_load_async_to_lds_b128(
      (__attribute__((address_space(1))) v4i_vs*)(unsigned long long)g,
      (__attribute__((address_space(3))) v4i_vs*)(unsigned)(unsigned long long)lds,
      0, 0);
#else
  *(v4u*)lds = *(const v4u*)g;
#endif
}

__device__ __forceinline__ void wait_async0() {
#if __has_builtin(__builtin_amdgcn_s_wait_asynccnt)
  __builtin_amdgcn_s_wait_asynccnt(0);
#else
  asm volatile("s_wait_asynccnt 0x0" ::: "memory");
#endif
}

// ---------------------------------------------------------------------------
// Cross-lane row reductions over each 16-lane half (wave32):
// xor1/2/4 via DPP8, xor8 via permlane16 -> pure VALU, no LDS.
// ---------------------------------------------------------------------------
template <int SEL>
__device__ __forceinline__ float dpp8f(float x) {
  return __builtin_bit_cast(
      float, __builtin_amdgcn_mov_dpp8(__builtin_bit_cast(int, x), SEL));
}
__device__ __forceinline__ float xor8f(float x) {
#if __has_builtin(__builtin_amdgcn_permlane16)
  unsigned u = __builtin_bit_cast(unsigned, x);
  unsigned r = __builtin_amdgcn_permlane16(u, u, 0xFEDCBA98u, 0x76543210u,
                                           false, false);
  return __builtin_bit_cast(float, r);
#else
  return __shfl_xor(x, 8, 32);
#endif
}
// DPP8 selector imms (octal digit per lane, lane7..lane0): i^1, i^2, i^4
#define DPP8_XOR1 067452301
#define DPP8_XOR2 054761032
#define DPP8_XOR4 032107654

__device__ __forceinline__ float rowmax16(float x) {
  x = fmaxf(x, dpp8f<DPP8_XOR1>(x));
  x = fmaxf(x, dpp8f<DPP8_XOR2>(x));
  x = fmaxf(x, dpp8f<DPP8_XOR4>(x));
  x = fmaxf(x, xor8f(x));
  return x;
}
__device__ __forceinline__ float rowsum16(float x) {
  x += dpp8f<DPP8_XOR1>(x);
  x += dpp8f<DPP8_XOR2>(x);
  x += dpp8f<DPP8_XOR4>(x);
  x += xor8f(x);
  return x;
}

// ---------------------------------------------------------------------------
// Weight pack: W[k][c] f32 -> Wt[c][k] f16  (contiguous-K rows for B-frags)
// ---------------------------------------------------------------------------
__global__ __launch_bounds__(256) void pack_w_kernel(
    const float* __restrict__ wqkv, const float* __restrict__ wout,
    _Float16* __restrict__ wtq, _Float16* __restrict__ wto) {
  int idx = blockIdx.x * 256 + threadIdx.x;
  const int t1 = QKVN * DIMC;
  if (idx < t1) {
    int c = idx / DIMC, k = idx % DIMC;
    wtq[idx] = (_Float16)wqkv[(size_t)k * QKVN + c];
  } else {
    int j = idx - t1;
    if (j < DIMC * DIMC) {
      int c = j / DIMC, k = j % DIMC;
      wto[j] = (_Float16)wout[(size_t)k * DIMC + c];
    }
  }
}

// ---------------------------------------------------------------------------
// LayerNorm: one row (768) per block of 256 threads, f16 output
// ---------------------------------------------------------------------------
__global__ __launch_bounds__(256) void ln_kernel(
    const float* __restrict__ x, const float* __restrict__ g,
    const float* __restrict__ bta, _Float16* __restrict__ xn) {
  int row = blockIdx.x;
  const float* xr = x + (size_t)row * DIMC;
  int t = threadIdx.x;
  float v0 = xr[t], v1 = xr[t + 256], v2 = xr[t + 512];
  float s = v0 + v1 + v2;
  float s2 = v0 * v0 + v1 * v1 + v2 * v2;
  __shared__ float red[32];
  #pragma unroll
  for (int m = 1; m < 32; m <<= 1) {
    s  += __shfl_xor(s,  m, 32);
    s2 += __shfl_xor(s2, m, 32);
  }
  int wid = t >> 5;
  if ((t & 31) == 0) { red[wid] = s; red[8 + wid] = s2; }
  __syncthreads();
  if (t == 0) {
    float a = 0.f, b = 0.f;
    #pragma unroll
    for (int i = 0; i < 8; i++) { a += red[i]; b += red[8 + i]; }
    red[16] = a; red[17] = b;
  }
  __syncthreads();
  float mean = red[16] * (1.0f / DIMC);
  float var  = red[17] * (1.0f / DIMC) - mean * mean;
  float inv  = rsqrtf(var + 1e-5f);
  _Float16* o = xn + (size_t)row * DIMC;
  o[t]       = (_Float16)((v0 - mean) * inv * g[t]       + bta[t]);
  o[t + 256] = (_Float16)((v1 - mean) * inv * g[t + 256] + bta[t + 256]);
  o[t + 512] = (_Float16)((v2 - mean) * inv * g[t + 512] + bta[t + 512]);
}

// ---------------------------------------------------------------------------
// Generic WMMA GEMM: C[8192 x NOUT] = A[8192 x 768] * Bt[NOUT x 768]^T
// Block tile 128(M) x 64(N), 8 waves each 32x32 (2x2 WMMA frags), K step 32.
// Double-buffered LDS (pitch 40 halves -> conflict-free ds_load_b128) filled
// by async global->LDS copies; one barrier per K step.
// MODE 0: scatter into Q / K / V^T (per-head).  MODE 1: f32 out + bias.
// ---------------------------------------------------------------------------
template <int NOUT, int MODE>
__global__ __launch_bounds__(256) void gemm_kernel(
    const _Float16* __restrict__ A, const _Float16* __restrict__ Bt,
    _Float16* __restrict__ Qb, _Float16* __restrict__ Kb,
    _Float16* __restrict__ Vtb,
    const float* __restrict__ bias, float* __restrict__ Out) {
  constexpr int KP = 40;
  __shared__ __attribute__((aligned(16))) _Float16 As[2][128 * KP];
  __shared__ __attribute__((aligned(16))) _Float16 Bs[2][64 * KP];

  const int m0 = blockIdx.y * 128;
  const int n0 = blockIdx.x * 64;
  const int tid = threadIdx.x;
  const int lane = tid & 31, w = tid >> 5;
  const int laneLo = lane & 15, hi = lane >> 4;
  const int moff = (w & 3) * 32, noff = (w >> 2) * 32;

  // per-thread cooperative-copy coordinates
  const int ar0 = (tid * 2) >> 2, ac0 = ((tid * 2) & 3) * 8;
  const int ar1 = (tid * 2 + 1) >> 2, ac1 = ((tid * 2 + 1) & 3) * 8;
  const int br = tid >> 2, bc = (tid & 3) * 8;

  auto issue_tiles = [&](int k0, int buf) {
    async_copy16(&As[buf][ar0 * KP + ac0],
                 &A[(size_t)(m0 + ar0) * DIMC + k0 + ac0]);
    async_copy16(&As[buf][ar1 * KP + ac1],
                 &A[(size_t)(m0 + ar1) * DIMC + k0 + ac1]);
    async_copy16(&Bs[buf][br * KP + bc],
                 &Bt[(size_t)(n0 + br) * DIMC + k0 + bc]);
  };

  FAcc acc[2][2];
  #pragma unroll
  for (int i = 0; i < 2; i++)
    #pragma unroll
    for (int j = 0; j < 2; j++)
      #pragma unroll
      for (int r = 0; r < 8; r++) acc[i][j].f[r] = 0.f;

  constexpr int NKT = DIMC / 32;   // 24
  issue_tiles(0, 0);
  wait_async0();
  __syncthreads();

  for (int kt = 0; kt < NKT; kt++) {
    const int cur = kt & 1;
    if (kt + 1 < NKT) issue_tiles((kt + 1) * 32, 1 - cur);

    HFrag a[2], b[2];
    #pragma unroll
    for (int i = 0; i < 2; i++) {
      int r = moff + i * 16 + laneLo;
      a[i].q[0] = *(const v4u*)&As[cur][r * KP + hi * 8];
      a[i].q[1] = *(const v4u*)&As[cur][r * KP + 16 + hi * 8];
    }
    #pragma unroll
    for (int j = 0; j < 2; j++) {
      int r = noff + j * 16 + laneLo;
      b[j].q[0] = *(const v4u*)&Bs[cur][r * KP + hi * 8];
      b[j].q[1] = *(const v4u*)&Bs[cur][r * KP + 16 + hi * 8];
    }
    #pragma unroll
    for (int i = 0; i < 2; i++)
      #pragma unroll
      for (int j = 0; j < 2; j++)
        acc[i][j].v = __builtin_amdgcn_wmma_f32_16x16x32_f16(
            false, a[i].h, false, b[j].h, (short)0, acc[i][j].v, false, false);

    wait_async0();
    __syncthreads();
  }

  // epilogue
  #pragma unroll
  for (int i = 0; i < 2; i++) {
    #pragma unroll
    for (int j = 0; j < 2; j++) {
      #pragma unroll
      for (int r = 0; r < 8; r++) {
        int m = m0 + moff + i * 16 + hi * 8 + r;
        int c = n0 + noff + j * 16 + laneLo;
        float val = acc[i][j].f[r];
        if (MODE == 0) {
          int which = c / DIMC;
          int cc = c % DIMC;
          int hh = cc / DHC;
          int d  = cc & (DHC - 1);
          int bb = m >> 10;       // /1024
          int n  = m & 1023;
          size_t hb = (size_t)(bb * NHEAD + hh);
          if (which == 0)
            Qb[(hb * NSEQ + n) * DHC + d] = (_Float16)val;
          else if (which == 1)
            Kb[(hb * NSEQ + n) * DHC + d] = (_Float16)val;
          else
            Vtb[(hb * DHC + d) * NSEQ + n] = (_Float16)val;   // V transposed
        } else {
          Out[(size_t)m * NOUT + c] = val + bias[c];
        }
      }
    }
  }
}

// ---------------------------------------------------------------------------
// Flash attention: grid (B*H, N/128); 8 waves, each owns 16 query rows.
// Double-buffered K/V^T tiles via async global->LDS; per 32-key block:
// 2 S-tiles (2 WMMA each), DPP-based online softmax, P via per-wave LDS
// transpose, 4 PV WMMAs into f32 accumulators.
// ---------------------------------------------------------------------------
__global__ __launch_bounds__(256) void attn_kernel(
    const _Float16* __restrict__ Qb, const _Float16* __restrict__ Kb,
    const _Float16* __restrict__ Vtb, const float* __restrict__ rpb,
    _Float16* __restrict__ Oa) {
  constexpr int KTP = 72;   // K tile pitch (64+8 halves) -> 144B rows
  constexpr int VTP = 40;   // V^T / P tile pitch (32+8 halves) -> 80B rows
  __shared__ __attribute__((aligned(16))) _Float16 Ks[2][32 * KTP];   // [j][d]
  __shared__ __attribute__((aligned(16))) _Float16 Vs[2][64 * VTP];   // [d][j]
  __shared__ __attribute__((aligned(16))) _Float16 Ps[8 * 16 * VTP];  // per-wave

  const int bh = blockIdx.x;           // b*12 + h
  const int b = bh / NHEAD, h = bh % NHEAD;
  const int tid = threadIdx.x, lane = tid & 31, w = tid >> 5;
  const int laneLo = lane & 15, hi = lane >> 4;
  const int i0 = blockIdx.y * 128 + w * 16;   // query row base for this wave

  const _Float16* qbase = Qb + (size_t)bh * NSEQ * DHC;
  const _Float16* kbase = Kb + (size_t)bh * NSEQ * DHC;
  const _Float16* vbase = Vtb + (size_t)bh * DHC * NSEQ;
  const float* rpbh = rpb + (size_t)h * NSEQ * NSEQ;
  _Float16* Pw = &Ps[w * 16 * VTP];

  const int kr = tid >> 3, kc = (tid & 7) * 8;   // K-tile copy coords
  const int vr = tid >> 2, vc = (tid & 3) * 8;   // V-tile copy coords
  auto issue_kv = [&](int j0, int buf) {
    async_copy16(&Ks[buf][kr * KTP + kc],
                 &kbase[(size_t)(j0 + kr) * DHC + kc]);
    async_copy16(&Vs[buf][vr * VTP + vc],
                 &vbase[(size_t)vr * NSEQ + j0 + vc]);
  };

  // Q fragments held in registers for the whole kernel (K = d, 2 chunks of 32)
  HFrag aq[2];
  #pragma unroll
  for (int kk = 0; kk < 2; kk++) {
    const _Float16* qr = qbase + (size_t)(i0 + laneLo) * DHC + kk * 32 + hi * 8;
    aq[kk].q[0] = *(const v4u*)qr;
    aq[kk].q[1] = *(const v4u*)(qr + 16);
  }

  FAcc acc[4];
  float m8[8], l8[8];
  #pragma unroll
  for (int r = 0; r < 8; r++) { m8[r] = -__builtin_inff(); l8[r] = 0.f; }
  #pragma unroll
  for (int ch = 0; ch < 4; ch++)
    #pragma unroll
    for (int r = 0; r < 8; r++) acc[ch].f[r] = 0.f;

  constexpr int NJB = NSEQ / 32;   // 32 key blocks
  issue_kv(0, 0);
  wait_async0();
  __syncthreads();

  for (int jb = 0; jb < NJB; jb++) {
    const int cur = jb & 1;
    const int j0 = jb * 32;
    if (jb + 1 < NJB) issue_kv(j0 + 32, 1 - cur);

    // S tiles: cols j0+t*16, K = 64 (two 16x16x32 WMMAs each)
    FAcc S[2];
    #pragma unroll
    for (int t = 0; t < 2; t++) {
      HFrag bk[2];
      #pragma unroll
      for (int kk = 0; kk < 2; kk++) {
        int jr = t * 16 + laneLo;
        bk[kk].q[0] = *(const v4u*)&Ks[cur][jr * KTP + kk * 32 + hi * 8];
        bk[kk].q[1] = *(const v4u*)&Ks[cur][jr * KTP + kk * 32 + 16 + hi * 8];
      }
      v8f s;
      #pragma unroll
      for (int r = 0; r < 8; r++) s[r] = 0.f;
      s = __builtin_amdgcn_wmma_f32_16x16x32_f16(false, aq[0].h, false,
                                                 bk[0].h, (short)0, s, false, false);
      s = __builtin_amdgcn_wmma_f32_16x16x32_f16(false, aq[1].h, false,
                                                 bk[1].h, (short)0, s, false, false);
      S[t].v = s;
      const float* rp = rpbh + (size_t)(i0 + hi * 8) * NSEQ + j0 + t * 16 + laneLo;
      #pragma unroll
      for (int r = 0; r < 8; r++) {
        S[t].f[r] = S[t].f[r] * SCALE_QK + rp[(size_t)r * NSEQ];
        if (t == 0 && j0 + 32 < NSEQ)   // prefetch next block's bias lines
          __builtin_prefetch(&rp[(size_t)r * NSEQ + 32], 0, 1);
      }
    }

    // online softmax per row (each row lives across one 16-lane half)
    float fscale[8];
    #pragma unroll
    for (int r = 0; r < 8; r++) {
      float tm = rowmax16(fmaxf(S[0].f[r], S[1].f[r]));
      float mn = fmaxf(m8[r], tm);
      fscale[r] = __expf(m8[r] - mn);
      float p0 = __expf(S[0].f[r] - mn);
      float p1 = __expf(S[1].f[r] - mn);
      float rs = rowsum16(p0 + p1);
      l8[r] = l8[r] * fscale[r] + rs;
      m8[r] = mn;
      int prow = r + hi * 8;
      Pw[prow * VTP + laneLo]      = (_Float16)p0;
      Pw[prow * VTP + 16 + laneLo] = (_Float16)p1;
    }
    #pragma unroll
    for (int ch = 0; ch < 4; ch++)
      #pragma unroll
      for (int r = 0; r < 8; r++) acc[ch].f[r] *= fscale[r];

    // P re-read in A-fragment layout (same-wave DS ops are in order)
    HFrag pf;
    pf.q[0] = *(const v4u*)&Pw[laneLo * VTP + hi * 8];
    pf.q[1] = *(const v4u*)&Pw[laneLo * VTP + 16 + hi * 8];

    #pragma unroll
    for (int ch = 0; ch < 4; ch++) {
      HFrag vf;
      int d = ch * 16 + laneLo;
      vf.q[0] = *(const v4u*)&Vs[cur][d * VTP + hi * 8];
      vf.q[1] = *(const v4u*)&Vs[cur][d * VTP + 16 + hi * 8];
      acc[ch].v = __builtin_amdgcn_wmma_f32_16x16x32_f16(
          false, pf.h, false, vf.h, (short)0, acc[ch].v, false, false);
    }

    wait_async0();
    __syncthreads();
  }

  // finalize: divide by row sums, store f16 [b][n][h*64+d]
  #pragma unroll
  for (int r = 0; r < 8; r++) {
    float inv = 1.0f / l8[r];
    int n = i0 + hi * 8 + r;
    #pragma unroll
    for (int ch = 0; ch < 4; ch++) {
      int col = h * DHC + ch * 16 + laneLo;
      Oa[((size_t)b * NSEQ + n) * DIMC + col] = (_Float16)(acc[ch].f[r] * inv);
    }
  }
}

// ---------------------------------------------------------------------------
extern "C" void kernel_launch(void* const* d_in, const int* in_sizes, int n_in,
                              void* d_out, int out_size, void* d_ws, size_t ws_size,
                              hipStream_t stream) {
  const float* x    = (const float*)d_in[0];
  const float* rpb  = (const float*)d_in[1];
  const float* Wqkv = (const float*)d_in[2];
  const float* Wout = (const float*)d_in[3];
  const float* bout = (const float*)d_in[4];
  const float* lng  = (const float*)d_in[5];
  const float* lnb  = (const float*)d_in[6];
  float* out = (float*)d_out;

  char* ws = (char*)d_ws;
  size_t off = 0;
  auto alloc = [&](size_t bytes) -> void* {
    void* p = ws + off;
    off += (bytes + 255) & ~(size_t)255;
    return p;
  };
  const size_t ROWS = (size_t)BATCH * NSEQ;               // 8192
  _Float16* xn   = (_Float16*)alloc(ROWS * DIMC * 2);     // 12.6 MB
  _Float16* wtq  = (_Float16*)alloc((size_t)QKVN * DIMC * 2);
  _Float16* wto  = (_Float16*)alloc((size_t)DIMC * DIMC * 2);
  _Float16* qb   = (_Float16*)alloc(ROWS * DIMC * 2);
  _Float16* kb   = (_Float16*)alloc(ROWS * DIMC * 2);
  _Float16* vtb  = (_Float16*)alloc(ROWS * DIMC * 2);
  _Float16* attn = (_Float16*)alloc(ROWS * DIMC * 2);
  (void)in_sizes; (void)n_in; (void)out_size; (void)ws_size;

  pack_w_kernel<<<9216, 256, 0, stream>>>(Wqkv, Wout, wtq, wto);
  ln_kernel<<<(int)ROWS, 256, 0, stream>>>(x, lng, lnb, xn);
  gemm_kernel<QKVN, 0><<<dim3(QKVN / 64, ROWS / 128), 256, 0, stream>>>(
      xn, wtq, qb, kb, vtb, nullptr, nullptr);
  attn_kernel<<<dim3(BATCH * NHEAD, NSEQ / 128), 256, 0, stream>>>(
      qb, kb, vtb, rpb, attn);
  gemm_kernel<DIMC, 1><<<dim3(DIMC / 64, ROWS / 128), 256, 0, stream>>>(
      attn, wto, nullptr, nullptr, nullptr, bout, out);
}
